// ResnetBlock_62775241998780
// MI455X (gfx1250) — matrix operand order repaired and checked by reading, not compiled, not run on hardware
//
#include <hip/hip_runtime.h>
#include <hip/hip_bf16.h>

// ---------------------------------------------------------------------------
// Problem constants (from reference)
// ---------------------------------------------------------------------------
constexpr int N_NODES = 131072;
constexpr int C       = 64;     // C_IN == C_OUT
constexpr int KNB     = 27;     // neighbor count
constexpr int B_BATCH = 8;
constexpr int EMB     = 512;
constexpr int GROUPS  = 32;     // channels-per-group = 2
constexpr float EPS   = 1e-5f;

constexpr int WPACK_HALVES = KNB * 8 * 32 * 16;        // 110592 halves
constexpr int WPACK_BYTES  = WPACK_HALVES * 2;         // 221184 B (fits 320KB LDS)

// WMMA vector types (probe-verified for gfx1250)
typedef __attribute__((ext_vector_type(16))) _Float16 v16h;
typedef __attribute__((ext_vector_type(8)))  _Float16 v8h;
typedef __attribute__((ext_vector_type(8)))  float    v8f;

// TDM descriptor group vector types (per CDNA5_HIP.md probe)
typedef __attribute__((ext_vector_type(4))) unsigned int u32x4;
typedef __attribute__((ext_vector_type(8))) int          i32x8;
typedef __attribute__((ext_vector_type(4))) int          i32x4;

union Frag32 { v16h v; v8h h[2]; };          // 32-byte A/B fragment per lane
union Half16 { _Float16 e[16]; v8h h[2]; };  // packing helper

static __device__ __forceinline__ float silu(float y) {
    return y * (1.0f / (1.0f + __expf(-y)));
}

// ---------------------------------------------------------------------------
// Zero the per-(batch,group) accumulators (must run every call: atomics).
// ---------------------------------------------------------------------------
__global__ void zero_stats(float* __restrict__ gsum, float* __restrict__ gss,
                           float* __restrict__ gcnt) {
    int tid = threadIdx.x;                  // 256 threads
    gsum[tid] = 0.0f;
    gss[tid]  = 0.0f;
    if (tid < B_BATCH) gcnt[tid] = 0.0f;
}

// ---------------------------------------------------------------------------
// Per-sample, per-group sums / sum-of-squares / counts.
// Thread = (node, quarter-row of 16 channels) -> 8 group partials.
// LDS f32 atomics within the block, then sparse global f32 atomics.
// ---------------------------------------------------------------------------
__global__ void __launch_bounds__(256)
gn_stats(const float* __restrict__ xin, const int* __restrict__ bid,
         float* __restrict__ gsum, float* __restrict__ gss,
         float* __restrict__ gcnt) {
    __shared__ float ls[B_BATCH * GROUPS];
    __shared__ float lq[B_BATCH * GROUPS];
    __shared__ float lc[B_BATCH];
    int tid = threadIdx.x;
    ls[tid] = 0.0f; lq[tid] = 0.0f;
    if (tid < B_BATCH) lc[tid] = 0.0f;
    __syncthreads();

    int gid = blockIdx.x * 256 + tid;       // N_NODES*4 threads total
    int n = gid >> 2, q = gid & 3;
    int b = bid[n];
    const float4* row = (const float4*)(xin + (size_t)n * C + q * 16);
    float4 vv[4] = {row[0], row[1], row[2], row[3]};

    float sv[8], qv[8];
#pragma unroll
    for (int i = 0; i < 4; ++i) {
        sv[2 * i]     = vv[i].x + vv[i].y;
        sv[2 * i + 1] = vv[i].z + vv[i].w;
        qv[2 * i]     = vv[i].x * vv[i].x + vv[i].y * vv[i].y;
        qv[2 * i + 1] = vv[i].z * vv[i].z + vv[i].w * vv[i].w;
    }
    int gbase = b * GROUPS + q * 8;
#pragma unroll
    for (int i = 0; i < 8; ++i) {
        atomicAdd(&ls[gbase + i], sv[i]);
        atomicAdd(&lq[gbase + i], qv[i]);
    }
    if (q == 0) atomicAdd(&lc[b], 1.0f);
    __syncthreads();

    if (ls[tid] != 0.0f) atomicAdd(&gsum[tid], ls[tid]);
    if (lq[tid] != 0.0f) atomicAdd(&gss[tid],  lq[tid]);
    if (tid < B_BATCH && lc[tid] != 0.0f) atomicAdd(&gcnt[tid], lc[tid]);
}

// ---------------------------------------------------------------------------
// mean / rstd per (batch, group)
// ---------------------------------------------------------------------------
__global__ void gn_finalize(const float* __restrict__ gsum,
                            const float* __restrict__ gss,
                            const float* __restrict__ gcnt,
                            float* __restrict__ mean, float* __restrict__ rstd) {
    int tid = threadIdx.x;                  // 256 = B*G
    int b = tid >> 5;
    float denom = gcnt[b] * 2.0f;           // cpg == 2
    float m = gsum[tid] / denom;
    float var = gss[tid] / denom - m * m;
    mean[tid] = m;
    rstd[tid] = rsqrtf(var + EPS);
}

// ---------------------------------------------------------------------------
// y = silu(gamma*(x-mean)*rstd + beta), cast to f16 (WMMA input precision).
// Thread = (node, quarter-row).
// ---------------------------------------------------------------------------
__global__ void __launch_bounds__(256)
gn_apply_silu(const float* __restrict__ xin, const int* __restrict__ bid,
              const float* __restrict__ mean, const float* __restrict__ rstd,
              const float* __restrict__ gamma, const float* __restrict__ beta,
              _Float16* __restrict__ hout) {
    int gid = blockIdx.x * 256 + threadIdx.x;
    int n = gid >> 2, q = gid & 3;
    int b = bid[n];
    const float4* row = (const float4*)(xin + (size_t)n * C + q * 16);
    Half16 u;
#pragma unroll
    for (int i = 0; i < 4; ++i) {
        float4 v = row[i];
        float vals[4] = {v.x, v.y, v.z, v.w};
#pragma unroll
        for (int j = 0; j < 4; ++j) {
            int c = q * 16 + i * 4 + j;
            int g = c >> 1;
            float m = mean[b * GROUPS + g];
            float r = rstd[b * GROUPS + g];
            float y = (vals[j] - m) * r * gamma[c] + beta[c];
            u.e[i * 4 + j] = (_Float16)silu(y);
        }
    }
    v8h* dst = (v8h*)(hout + (size_t)n * C + q * 16);
    dst[0] = u.h[0];
    dst[1] = u.h[1];
}

// ---------------------------------------------------------------------------
// Repack W[k][cin][cout] (f32) into WMMA B-fragment order (f16).
// Fragment (k, chunk, t): lane<16 -> col o=t*16+lane, K rows chunk*32+{0..7,16..23};
// lane>=16 -> same cols, K rows chunk*32+{8..15,24..31}   (ISA 7.12.2 B striping).
// ---------------------------------------------------------------------------
__global__ void repack_w(const float* __restrict__ W, _Float16* __restrict__ wp) {
    int k    = blockIdx.x;          // 27
    int tid  = threadIdx.x;         // 256 = 8 frags * 32 lanes
    int lane = tid & 31;
    int ft   = tid >> 5;            // chunk*4 + t
    int chunk = ft >> 2, t = ft & 3;
    int o     = t * 16 + (lane & 15);
    int cbase = chunk * 32 + ((lane < 16) ? 0 : 8);
    Half16 u;
#pragma unroll
    for (int h = 0; h < 16; ++h)
        u.e[h] = (_Float16)W[((size_t)k * C + cbase + (h < 8 ? h : h + 8)) * C + o];
    v8h* dst = (v8h*)(wp + (((size_t)k * 8 + ft) * 32 + lane) * 16);
    dst[0] = u.h[0];
    dst[1] = u.h[1];
}

// ---------------------------------------------------------------------------
// time embedding: silu(time_emb) @ Wt + bt   (tiny: 8x512x64)
// ---------------------------------------------------------------------------
__global__ void time_mlp(const float* __restrict__ temb_in,
                         const float* __restrict__ Wt,
                         const float* __restrict__ bt,
                         float* __restrict__ tout) {
    int b = blockIdx.x, o = threadIdx.x;    // 8 x 64
    float acc = bt[o];
    for (int e = 0; e < EMB; ++e)
        acc += silu(temb_in[b * EMB + e]) * Wt[e * C + o];
    tout[b * C + o] = acc;
}

// ---------------------------------------------------------------------------
// Stage the full repacked weight set (221184 B) into LDS, once per block.
// Preferred path: Tensor Data Mover (single tensor_load_to_lds issued by
// wave 0, 1-D tile: data_size=8B, tile_dim0=tensor_dim0=27648=0x6C00),
// waited with s_wait_tensorcnt.  Fallback: cooperative b128 copy.
// ---------------------------------------------------------------------------
static __device__ __forceinline__ void stage_weights_to_lds(
        const _Float16* __restrict__ wp, _Float16* __restrict__ wlds) {
#if defined(__has_builtin) && __has_builtin(__builtin_amdgcn_tensor_load_to_lds)
    if ((threadIdx.x >> 5) == 0) {          // one wave issues the TDM op
        unsigned long long ga = (unsigned long long)wp;
        // D# Group 0: count=1 | lds_addr=0 | global_addr | type=2
        u32x4 g0 = { 1u,
                     0u,
                     (unsigned)(ga & 0xFFFFFFFFu),
                     (unsigned)((ga >> 32) & 0x01FFFFFFu) | (2u << 30) };
        // D# Group 1: data_size=3 (8B); tensor_dim0=tile_dim0=0x6C00 (27648
        // 8B elems = 221184B); tensor_dim1=1, tile_dim1=1; stride0=0x6C00.
        i32x8 g1 = { (int)(3u << 16),        // wg_mask=0, data_size=3
                     (int)0x6C000000,        // tensor_dim0[15:0] << 16
                     (int)0x00010000,        // tensor_dim0 hi=0 | tensor_dim1 lo=1
                     (int)0x6C000000,        // tensor_dim1 hi=0 | tile_dim0=0x6C00
                     (int)0x00000001,        // tile_dim1=1 | tile_dim2=0
                     (int)0x00006C00,        // tensor_dim0_stride lo
                     0, 0 };
        i32x4 gz = { 0, 0, 0, 0 };
#if __clang_major__ >= 23
        i32x8 gz8 = { 0, 0, 0, 0, 0, 0, 0, 0 };
        __builtin_amdgcn_tensor_load_to_lds(g0, g1, gz, gz, gz8, 0);
#else
        __builtin_amdgcn_tensor_load_to_lds(g0, g1, gz, gz, 0);
#endif
#if __has_builtin(__builtin_amdgcn_s_wait_tensorcnt)
        __builtin_amdgcn_s_wait_tensorcnt(0);
#else
        asm volatile("s_wait_tensorcnt 0x0" ::: "memory");
#endif
    }
#else
    // Cooperative fallback: 13824 x 16B
    const float4* src = (const float4*)wp;
    float4*       dst = (float4*)wlds;
    for (int i = threadIdx.x; i < WPACK_BYTES / 16; i += blockDim.x)
        dst[i] = src[i];
#endif
    __syncthreads();
}

// ---------------------------------------------------------------------------
// Gathered conv as WMMA GEMM. One wave per 16-node tile, 16 waves/block.
// A (16x32 f16) gathered per-lane from global per ISA 7.12.2 A layout:
//   lane<16: row m=lane, K {0..7, 16..23}; lane>=16: row m=lane-16, K {8..15, 24..31}
// B fragments come from LDS (staged once per block).
// 27 neighbors x 2 K-chunks x 4 N-tiles = 216 v_wmma_f32_16x16x32_f16 / tile.
// Epilogue folds bias, scattered time-embedding, residual (C/D layout 7.12.2).
// ---------------------------------------------------------------------------
__global__ void __launch_bounds__(512)
conv_wmma(const _Float16* __restrict__ hin, const int* __restrict__ neigh,
          const _Float16* __restrict__ wp, const float* __restrict__ bias,
          const float* __restrict__ temb, const int* __restrict__ bid,
          const float* __restrict__ resid, float* __restrict__ out) {
    extern __shared__ _Float16 wlds[];      // WPACK_BYTES dynamic LDS

    stage_weights_to_lds(wp, wlds);

    const int lane = threadIdx.x & 31;
    const int wave = threadIdx.x >> 5;      // 0..15
    const int tile = blockIdx.x * 16 + wave;
    const int base = tile * 16;
    const int m    = lane & 15;
    const int hi   = lane >> 4;             // 0 or 1

    v8f acc[4] = {};                        // four 16x16 f32 accumulators

    const int* nrow = neigh + (size_t)(base + m) * KNB;
    int idx = nrow[0];
    for (int k = 0; k < KNB; ++k) {
        const _Float16* row = hin + (size_t)idx * C;
        if (k + 1 < KNB) {
            idx = nrow[k + 1];
            __builtin_prefetch(hin + (size_t)idx * C, 0, 0);  // global_prefetch_b8
        }
#pragma unroll
        for (int chunk = 0; chunk < 2; ++chunk) {
            Frag32 a;
            a.h[0] = *(const v8h*)(row + chunk * 32 + hi * 8);
            a.h[1] = *(const v8h*)(row + chunk * 32 + 16 + hi * 8);
            const v8h* wb =
                (const v8h*)(wlds + (((size_t)k * 8 + chunk * 4) * 32 + lane) * 16);
#pragma unroll
            for (int t = 0; t < 4; ++t) {
                Frag32 bf;
                bf.h[0] = wb[t * 64 + 0];   // t*32lanes*16halves = t*64 v8h (ds_load_b128)
                bf.h[1] = wb[t * 64 + 1];
                acc[t] = __builtin_amdgcn_wmma_f32_16x16x32_f16(
                    false, a.v, false, bf.v, (short)0, acc[t], false, false);
            }
        }
    }

    // Epilogue: C/D layout -> VGPR j: row = base + hi*8 + j, col = t*16 + m
#pragma unroll
    for (int j = 0; j < 8; ++j) {
        const int n  = base + hi * 8 + j;
        const int bb = temb ? bid[n] : 0;
#pragma unroll
        for (int t = 0; t < 4; ++t) {
            const int o = t * 16 + m;
            float v = acc[t][j] + bias[o];
            if (temb)  v += temb[bb * C + o];
            if (resid) v += resid[(size_t)n * C + o];
            out[(size_t)n * C + o] = v;
        }
    }
}

// ---------------------------------------------------------------------------
// Host launcher
// ---------------------------------------------------------------------------
extern "C" void kernel_launch(void* const* d_in, const int* in_sizes, int n_in,
                              void* d_out, int out_size, void* d_ws, size_t ws_size,
                              hipStream_t stream) {
    (void)in_sizes; (void)n_in; (void)out_size; (void)ws_size;

    const float* x        = (const float*)d_in[0];
    const float* time_emb = (const float*)d_in[1];
    const int*   batch_id = (const int*)d_in[2];
    const int*   neigh    = (const int*)d_in[3];
    const float* g1       = (const float*)d_in[4];
    const float* be1      = (const float*)d_in[5];
    const float* W1       = (const float*)d_in[6];
    const float* b1       = (const float*)d_in[7];
    const float* Wt       = (const float*)d_in[8];
    const float* bt       = (const float*)d_in[9];
    const float* g2       = (const float*)d_in[10];
    const float* be2      = (const float*)d_in[11];
    const float* W2       = (const float*)d_in[12];
    const float* b2       = (const float*)d_in[13];
    float* out = (float*)d_out;

    // Workspace carve-out (256B aligned)
    char* ws = (char*)d_ws;
    size_t off = 0;
    auto carve = [&](size_t bytes) -> void* {
        void* p = ws + off;
        off = (off + bytes + 255) & ~(size_t)255;
        return p;
    };
    _Float16* h16  = (_Float16*)carve((size_t)N_NODES * C * 2);   // 16 MB (f16 activations)
    float*    h1   = (float*)   carve((size_t)N_NODES * C * 4);   // 32 MB (conv1 out, f32)
    _Float16* wp1  = (_Float16*)carve(WPACK_BYTES);
    _Float16* wp2  = (_Float16*)carve(WPACK_BYTES);
    float*    tbuf = (float*)   carve((size_t)B_BATCH * C * 4);
    float*    gsum = (float*)   carve(B_BATCH * GROUPS * 4);
    float*    gss  = (float*)   carve(B_BATCH * GROUPS * 4);
    float*    gcnt = (float*)   carve(B_BATCH * 4);
    float*    gmu  = (float*)   carve(B_BATCH * GROUPS * 4);
    float*    grs  = (float*)   carve(B_BATCH * GROUPS * 4);

    const int statsGrid = (N_NODES * 4) / 256;       // 2048
    const int convGrid  = (N_NODES / 16) / 16;       // 512 blocks x 16 waves
    const size_t convLds = WPACK_BYTES;              // 221184 B dynamic LDS

    // Weight repack + time MLP (independent of activations)
    repack_w<<<KNB, 256, 0, stream>>>(W1, wp1);
    repack_w<<<KNB, 256, 0, stream>>>(W2, wp2);
    time_mlp<<<B_BATCH, C, 0, stream>>>(time_emb, Wt, bt, tbuf);

    // --- GN1 -> SiLU -> f16 ---
    zero_stats<<<1, 256, 0, stream>>>(gsum, gss, gcnt);
    gn_stats<<<statsGrid, 256, 0, stream>>>(x, batch_id, gsum, gss, gcnt);
    gn_finalize<<<1, 256, 0, stream>>>(gsum, gss, gcnt, gmu, grs);
    gn_apply_silu<<<statsGrid, 256, 0, stream>>>(x, batch_id, gmu, grs, g1, be1, h16);

    // --- conv1 (+bias +time-emb) ---
    conv_wmma<<<convGrid, 512, convLds, stream>>>(h16, neigh, wp1, b1, tbuf,
                                                  batch_id, nullptr, h1);

    // --- GN2 -> SiLU -> f16 ---
    zero_stats<<<1, 256, 0, stream>>>(gsum, gss, gcnt);
    gn_stats<<<statsGrid, 256, 0, stream>>>(h1, batch_id, gsum, gss, gcnt);
    gn_finalize<<<1, 256, 0, stream>>>(gsum, gss, gcnt, gmu, grs);
    gn_apply_silu<<<statsGrid, 256, 0, stream>>>(h1, batch_id, gmu, grs, g2, be2, h16);

    // --- conv2 (+bias +residual x) ---
    conv_wmma<<<convGrid, 512, convLds, stream>>>(h16, neigh, wp2, b2, nullptr,
                                                  batch_id, x, out);
}
